// AID_80367428043393
// MI455X (gfx1250) — matrix operand order repaired and checked by reading, not compiled, not run on hardware
//
#include <hip/hip_runtime.h>
#include <hip/hip_bf16.h>

// ---------------------------------------------------------------------------
// Types
// ---------------------------------------------------------------------------
typedef __attribute__((ext_vector_type(16))) __bf16        v16bf;
typedef __attribute__((ext_vector_type(8)))  float         v8f;
typedef __attribute__((ext_vector_type(4)))  unsigned int  u32x4;

union FragBF16 {
    v16bf v;
    u32x4 q[2];
    unsigned short u[16];
};

__device__ __forceinline__ unsigned short f32_to_bf16_bits(float f) {
    unsigned int u = __float_as_uint(f);
    u += 0x7FFFu + ((u >> 16) & 1u);   // round-to-nearest-even
    return (unsigned short)(u >> 16);
}

__device__ __forceinline__ float featf(float x) {
    // elu(x)+1 == (x>0 ? x+1 : exp(x))
    return x > 0.0f ? x + 1.0f : __expf(x);
}

// A fragment (16x32 bf16, ISA 7.12.2): lanes 0-15 hold K {k0..k0+7, k0+16..k0+23},
// lanes 16-31 hold K {k0+8..k0+15, k0+24..k0+31}.
__device__ __forceinline__ void load_frag_a(FragBF16& f, const unsigned short* row,
                                            int k0, int halfSel) {
    f.q[0] = *reinterpret_cast<const u32x4*>(row + k0 + halfSel * 8);
    f.q[1] = *reinterpret_cast<const u32x4*>(row + k0 + 16 + halfSel * 8);
}

// B fragment (32x16 bf16): lane = column N; lanes 0-15 carry K {k0..k0+7, k0+16..k0+23},
// lanes 16-31 carry K {k0+8..k0+15, k0+24..k0+31} (p already offset by halfSel*8).
__device__ __forceinline__ void load_frag_b(FragBF16& f, const unsigned short* p) {
    f.q[0] = *reinterpret_cast<const u32x4*>(p);
    f.q[1] = *reinterpret_cast<const u32x4*>(p + 16);
}

// ---------------------------------------------------------------------------
// Weight convert + transpose: W[K][N] (f32) -> Wt[N][K] (bf16)
// ---------------------------------------------------------------------------
__global__ void aid_convT_bf16(const float* __restrict__ W,
                               unsigned short* __restrict__ Wt,
                               int K, int Ncol) {
    size_t i = (size_t)blockIdx.x * 256 + threadIdx.x;
    if (i >= (size_t)K * Ncol) return;
    int k = (int)(i / Ncol);
    int n = (int)(i % Ncol);
    Wt[(size_t)n * K + k] = f32_to_bf16_bits(W[i]);
}

// ---------------------------------------------------------------------------
// Generic WMMA GEMM:  C[M,Nout] = act(A[M,K] @ Wt^T + bias)
//   A  : bf16 row-major [M,K];  Wt : bf16 row-major [Nout,K]
//   outF (opt): f32 at outF[m*ldo + coff + n];  outB (opt): bf16 [M,Nout]
//   act: 0 none, 1 relu, 2 feat (elu+1)
// Block: 128 threads = 4 waves; each wave computes a 32(M) x 64(N) strip
// (8 WMMA accumulators). K-loop is software-pipelined with register
// double-buffering so fragment loads overlap the WMMA bursts.
// Requirements: M % 128 == 0, K % 64 == 0, Nout % 64 == 0.
// ---------------------------------------------------------------------------
__global__ __launch_bounds__(128)
void aid_gemm_wmma(const unsigned short* __restrict__ A,
                   const unsigned short* __restrict__ Wt,
                   const float* __restrict__ bias,
                   float* __restrict__ outF, int ldo, int coff,
                   unsigned short* __restrict__ outB,
                   int M, int K, int Nout, int act) {
    const int lane    = threadIdx.x & 31;
    const int wave    = threadIdx.x >> 5;
    const int mbase   = (blockIdx.x * 4 + wave) * 32;
    const int ntile   = blockIdx.y * 64;
    const int halfSel = lane >> 4;     // 0: lanes 0-15, 1: lanes 16-31
    const int l15     = lane & 15;
    if (mbase >= M) return;            // wave-uniform (M % 128 == 0 per block)

    const v8f zero = {0, 0, 0, 0, 0, 0, 0, 0};
    v8f acc[2][4];
#pragma unroll
    for (int mi = 0; mi < 2; ++mi)
#pragma unroll
        for (int t = 0; t < 4; ++t) acc[mi][t] = zero;

    const unsigned short* Arow0 = A + (size_t)(mbase + l15) * K;
    const unsigned short* Arow1 = A + (size_t)(mbase + 16 + l15) * K;
    const unsigned short* Bp[4];
#pragma unroll
    for (int t = 0; t < 4; ++t)
        Bp[t] = Wt + (size_t)(ntile + t * 16 + l15) * K + halfSel * 8;

    FragBF16 a0c, a1c, bc[4];          // current set
    FragBF16 a0n, a1n, bn[4];          // next (prefetched) set

    // prologue: k0 = 0
    load_frag_a(a0c, Arow0, 0, halfSel);
    load_frag_a(a1c, Arow1, 0, halfSel);
#pragma unroll
    for (int t = 0; t < 4; ++t) load_frag_b(bc[t], Bp[t]);

    for (int k0 = 0; k0 < K; k0 += 64) {
        // prefetch k0+32 (always valid: K % 64 == 0)
        load_frag_a(a0n, Arow0, k0 + 32, halfSel);
        load_frag_a(a1n, Arow1, k0 + 32, halfSel);
#pragma unroll
        for (int t = 0; t < 4; ++t) load_frag_b(bn[t], Bp[t] + k0 + 32);

#pragma unroll
        for (int t = 0; t < 4; ++t) {
            acc[0][t] = __builtin_amdgcn_wmma_f32_16x16x32_bf16(
                false, a0c.v, false, bc[t].v, (short)0, acc[0][t], false, false);
            acc[1][t] = __builtin_amdgcn_wmma_f32_16x16x32_bf16(
                false, a1c.v, false, bc[t].v, (short)0, acc[1][t], false, false);
        }

        if (k0 + 64 < K) {             // prefetch k0+64 into current set
            load_frag_a(a0c, Arow0, k0 + 64, halfSel);
            load_frag_a(a1c, Arow1, k0 + 64, halfSel);
#pragma unroll
            for (int t = 0; t < 4; ++t) load_frag_b(bc[t], Bp[t] + k0 + 64);
        }

#pragma unroll
        for (int t = 0; t < 4; ++t) {
            acc[0][t] = __builtin_amdgcn_wmma_f32_16x16x32_bf16(
                false, a0n.v, false, bn[t].v, (short)0, acc[0][t], false, false);
            acc[1][t] = __builtin_amdgcn_wmma_f32_16x16x32_bf16(
                false, a1n.v, false, bn[t].v, (short)0, acc[1][t], false, false);
        }
    }

    // C/D layout: lanes 0-15 -> (M = mt+r, N), lanes 16-31 -> (M = mt+8+r, N)
#pragma unroll
    for (int mi = 0; mi < 2; ++mi) {
#pragma unroll
        for (int t = 0; t < 4; ++t) {
            const int n = ntile + t * 16 + l15;
            const float bv = bias ? bias[n] : 0.0f;
#pragma unroll
            for (int r = 0; r < 8; ++r) {
                const int m = mbase + mi * 16 + halfSel * 8 + r;
                float vv = acc[mi][t][r] + bv;
                if (act == 1)      vv = fmaxf(vv, 0.0f);
                else if (act == 2) vv = featf(vv);
                if (outF) outF[(size_t)m * ldo + coff + n] = vv;
                if (outB) outB[(size_t)m * Nout + n] = f32_to_bf16_bits(vv);
            }
        }
    }
}

// ---------------------------------------------------------------------------
// Row LayerNorm -> bf16 output. One block (256 thr) per row of width D.
// ---------------------------------------------------------------------------
__global__ __launch_bounds__(256)
void aid_layernorm_bf16(const float* __restrict__ x,
                        const float* __restrict__ g,
                        const float* __restrict__ b,
                        unsigned short* __restrict__ outB, int D) {
    const int row = blockIdx.x;
    const float* xr = x + (size_t)row * D;
    __shared__ float rsum[8], rsq[8];
    __shared__ float smean, sinv;

    float s = 0.0f, sq = 0.0f;
    for (int i = threadIdx.x; i < D; i += 256) {
        float v = xr[i];
        s += v; sq += v * v;
    }
    for (int o = 16; o > 0; o >>= 1) {
        s  += __shfl_down(s, o);
        sq += __shfl_down(sq, o);
    }
    if ((threadIdx.x & 31) == 0) {
        rsum[threadIdx.x >> 5] = s;
        rsq[threadIdx.x >> 5]  = sq;
    }
    __syncthreads();
    if (threadIdx.x == 0) {
        float ts = 0.0f, tq = 0.0f;
        for (int i = 0; i < 8; ++i) { ts += rsum[i]; tq += rsq[i]; }
        float m   = ts / D;
        float var = tq / D - m * m;
        smean = m;
        sinv  = rsqrtf(var + 1e-5f);
    }
    __syncthreads();
    for (int i = threadIdx.x; i < D; i += 256)
        outB[(size_t)row * D + i] =
            f32_to_bf16_bits((xr[i] - smean) * sinv * g[i] + b[i]);
}

// ---------------------------------------------------------------------------
// Elementwise helpers
// ---------------------------------------------------------------------------
__global__ void aid_copy_f32_bf16(const float* __restrict__ src,
                                  float* __restrict__ dstF,
                                  unsigned short* __restrict__ dstB, size_t n) {
    size_t i = (size_t)blockIdx.x * 256 + threadIdx.x;
    if (i >= n) return;
    float v = src[i];
    dstF[i] = v;
    dstB[i] = f32_to_bf16_bits(v);
}

// qq = feat((qq_pre + kv0) * E^-0.5), in place
__global__ void aid_qq_feat(float* __restrict__ qq,
                            const float* __restrict__ kv0, size_t n) {
    size_t i = (size_t)blockIdx.x * 256 + threadIdx.x;
    if (i >= n) return;
    qq[i] = featf((qq[i] + kv0[i]) * 0.125f);
}

// slots += h2 * (1/E);  slots_bf16 = bf16(slots)
__global__ void aid_slots_update(float* __restrict__ slots,
                                 unsigned short* __restrict__ slotsB,
                                 const float* __restrict__ h2, size_t n) {
    size_t i = (size_t)blockIdx.x * 256 + threadIdx.x;
    if (i >= n) return;
    float v = slots[i] + h2[i] * (1.0f / 64.0f);
    slots[i] = v;
    slotsB[i] = f32_to_bf16_bits(v);
}

// cat[r][0:64] = kv0 row r, cat[r][64:128] = slots row r (rows of width 64)
__global__ void aid_concat_bf16(const float* __restrict__ kv0,
                                const float* __restrict__ slots,
                                unsigned short* __restrict__ cat, size_t n) {
    size_t i = (size_t)blockIdx.x * 256 + threadIdx.x;
    if (i >= n) return;
    size_t r = i >> 7;
    int    c = (int)(i & 127);
    float v = (c < 64) ? kv0[r * 64 + c] : slots[r * 64 + (c - 64)];
    cat[i] = f32_to_bf16_bits(v);
}

__global__ void aid_add_f32(const float* __restrict__ a,
                            const float* __restrict__ b,
                            float* __restrict__ out, size_t n) {
    size_t i = (size_t)blockIdx.x * 256 + threadIdx.x;
    if (i >= n) return;
    out[i] = a[i] + b[i];
}

// ---------------------------------------------------------------------------
// Per-token slot competitive attention step.
//   kk,qq,vv : f32 [tokens, 32, 64]
//   h_ln     : bf16 [tokens*32, 64]  (LayerNorm(upd) with lnp_g/lnp_b)
// One block (256 thr) per token; all tiles live in LDS.
// ---------------------------------------------------------------------------
__global__ __launch_bounds__(256)
void aid_slot_attn(const float* __restrict__ kk,
                   const float* __restrict__ qq,
                   const float* __restrict__ vv,
                   const float* __restrict__ lnp_g,
                   const float* __restrict__ lnp_b,
                   unsigned short* __restrict__ h_ln) {
    const int token = blockIdx.x;
    __shared__ float skk[2048], sqq[2048], svv[2048], supd[2048];
    __shared__ float slog[1024];
    __shared__ float scol[32], smean[32], sinv[32];

    const size_t base = (size_t)token * 2048;
    for (int i = threadIdx.x; i < 2048; i += 256) {
        skk[i] = kk[base + i];
        sqq[i] = qq[base + i];
        svv[i] = vv[base + i];
    }
    __syncthreads();

    // logits[w][n] = sum_e kk[w,e] * qq[n,e]
    for (int idx = threadIdx.x; idx < 1024; idx += 256) {
        int w = idx >> 5, n = idx & 31;
        float s = 0.0f;
        for (int e = 0; e < 64; ++e) s += skk[w * 64 + e] * sqq[n * 64 + e];
        slog[idx] = s;
    }
    __syncthreads();

    // softmax over n per row w, then + 1e-6
    if (threadIdx.x < 32) {
        int w = threadIdx.x;
        float mx = -1e30f;
        for (int n = 0; n < 32; ++n) mx = fmaxf(mx, slog[w * 32 + n]);
        float sum = 0.0f;
        for (int n = 0; n < 32; ++n) {
            float e = __expf(slog[w * 32 + n] - mx);
            slog[w * 32 + n] = e;
            sum += e;
        }
        float inv = 1.0f / sum;
        for (int n = 0; n < 32; ++n) slog[w * 32 + n] = slog[w * 32 + n] * inv + 1e-6f;
    }
    __syncthreads();

    // slot competition: normalize over w (axis=2)
    if (threadIdx.x < 32) {
        int n = threadIdx.x;
        float s = 0.0f;
        for (int w = 0; w < 32; ++w) s += slog[w * 32 + n];
        scol[n] = 1.0f / s;
    }
    __syncthreads();

    // upd[n][e] = sum_w attn[w,n] * vv[w,e]
    for (int idx = threadIdx.x; idx < 2048; idx += 256) {
        int n = idx >> 6, e = idx & 63;
        float s = 0.0f;
        for (int w = 0; w < 32; ++w) s += slog[w * 32 + n] * svv[w * 64 + e];
        supd[idx] = s * scol[n];
    }
    __syncthreads();

    // per-slot LayerNorm stats over E=64
    if (threadIdx.x < 32) {
        int n = threadIdx.x;
        float m = 0.0f;
        for (int e = 0; e < 64; ++e) m += supd[n * 64 + e];
        m *= (1.0f / 64.0f);
        float vr = 0.0f;
        for (int e = 0; e < 64; ++e) {
            float d = supd[n * 64 + e] - m;
            vr += d * d;
        }
        vr *= (1.0f / 64.0f);
        smean[n] = m;
        sinv[n]  = rsqrtf(vr + 1e-5f);
    }
    __syncthreads();

    for (int idx = threadIdx.x; idx < 2048; idx += 256) {
        int n = idx >> 6, e = idx & 63;
        float v = (supd[idx] - smean[n]) * sinv[n] * lnp_g[e] + lnp_b[e];
        h_ln[base + idx] = f32_to_bf16_bits(v);
    }
}

// ---------------------------------------------------------------------------
// Linear attention, collapsed (no mask => associativity):
//   KV[b,h,e1,e2] = sum_t feat(slotsP[b,t,h,e1]) * slotsP[b,t,16+h,e2]
//   ksum[b,h,e1]  = sum_t feat(slotsP[b,t,h,e1])
// One block (256 thr) per (b,h); slotsP is f32 [B,S,32,64].
// ---------------------------------------------------------------------------
__global__ __launch_bounds__(256)
void aid_kv_accum(const float* __restrict__ slotsP,
                  float* __restrict__ KV, float* __restrict__ ksum) {
    const int bh = blockIdx.x;      // b*16 + h
    const int b = bh >> 4, h = bh & 15;
    __shared__ float skf[64], sv2[64];
    float acc[16];
#pragma unroll
    for (int j = 0; j < 16; ++j) acc[j] = 0.0f;
    float ks = 0.0f;

    for (int t = 0; t < 1024; ++t) {
        const size_t tok = ((size_t)(b * 1024 + t)) * 2048;
        if (threadIdx.x < 64) {
            float kv = slotsP[tok + (size_t)h * 64 + threadIdx.x];
            skf[threadIdx.x] = featf(kv);
            sv2[threadIdx.x] = slotsP[tok + (size_t)(16 + h) * 64 + threadIdx.x];
        }
        __syncthreads();
#pragma unroll
        for (int j = 0; j < 16; ++j) {
            int idx = threadIdx.x * 16 + j;       // e1 = idx>>6, e2 = idx&63
            acc[j] += skf[idx >> 6] * sv2[idx & 63];
        }
        if (threadIdx.x < 64) ks += skf[threadIdx.x];
        __syncthreads();
    }
#pragma unroll
    for (int j = 0; j < 16; ++j)
        KV[(size_t)bh * 4096 + threadIdx.x * 16 + j] = acc[j];
    if (threadIdx.x < 64) ksum[bh * 64 + threadIdx.x] = ks;
}

// out[b,s,h,e2] = (qf[h,:] . KV[b,h,:,e2]) / (qf[h,:] . ksum[b,h,:] + 1e-5)
// with qf = feat(q * E^-0.5). One block per token; writes bf16 [4096,1024].
__global__ __launch_bounds__(256)
void aid_attn_apply(const float* __restrict__ q,
                    const float* __restrict__ KV,
                    const float* __restrict__ ksum,
                    unsigned short* __restrict__ attn_out) {
    const int token = blockIdx.x;   // b*1024 + s
    const int b = token >> 10;
    __shared__ float sqf[1024];
    __shared__ float sden[16];

    const float* qrow = q + (size_t)token * 1024;
    for (int i = threadIdx.x; i < 1024; i += 256)
        sqf[i] = featf(qrow[i] * 0.125f);
    __syncthreads();

    if (threadIdx.x < 16) {
        int h = threadIdx.x;
        const float* kp = ksum + (b * 16 + h) * 64;
        float s = 0.0f;
        for (int e = 0; e < 64; ++e) s += sqf[h * 64 + e] * kp[e];
        sden[h] = 1.0f / (s + 1e-5f);
    }
    __syncthreads();

    for (int idx = threadIdx.x; idx < 1024; idx += 256) {
        int h = idx >> 6, e2 = idx & 63;
        const float* kvp = KV + (size_t)(b * 16 + h) * 4096 + e2;
        float s = 0.0f;
        for (int e1 = 0; e1 < 64; ++e1) s += sqf[h * 64 + e1] * kvp[e1 * 64];
        attn_out[(size_t)token * 1024 + idx] = f32_to_bf16_bits(s * sden[h]);
    }
}

// ---------------------------------------------------------------------------
// Host launch
// ---------------------------------------------------------------------------
extern "C" void kernel_launch(void* const* d_in, const int* in_sizes, int n_in,
                              void* d_out, int out_size, void* d_ws, size_t ws_size,
                              hipStream_t stream) {
    (void)in_sizes; (void)n_in; (void)out_size; (void)ws_size;

    constexpr int Bq = 4, S = 1024, D = 1024, H = 16, E = 64;
    constexpr int TOK = Bq * S;          // 4096 tokens
    constexpr int SR  = TOK * 2 * H;     // 131072 slot rows

    const float* x     = (const float*)d_in[0];
    const float* ln1_g = (const float*)d_in[1];
    const float* ln1_b = (const float*)d_in[2];
    const float* ln2_g = (const float*)d_in[3];
    const float* ln2_b = (const float*)d_in[4];
    const float* Wq = (const float*)d_in[5];  const float* bq = (const float*)d_in[6];
    const float* Wk = (const float*)d_in[7];  const float* bk = (const float*)d_in[8];
    const float* Wv = (const float*)d_in[9];  const float* bv = (const float*)d_in[10];
    const float* Wf = (const float*)d_in[11]; const float* bf = (const float*)d_in[12];
    const float* Wr1 = (const float*)d_in[13]; const float* br1 = (const float*)d_in[14];
    const float* Wr2 = (const float*)d_in[15]; const float* br2 = (const float*)d_in[16];
    const float* Wip = (const float*)d_in[17]; const float* bip = (const float*)d_in[18];
    const float* Wtq = (const float*)d_in[19]; const float* btq = (const float*)d_in[20];
    const float* Wtk = (const float*)d_in[21]; const float* btk = (const float*)d_in[22];
    const float* Wtv = (const float*)d_in[23]; const float* btv = (const float*)d_in[24];
    const float* lnp_g = (const float*)d_in[25]; const float* lnp_b = (const float*)d_in[26];
    const float* Wl1 = (const float*)d_in[27]; const float* bl1 = (const float*)d_in[28];
    const float* Wl2 = (const float*)d_in[29]; const float* bl2 = (const float*)d_in[30];
    const float* Wsp = (const float*)d_in[31]; const float* bsp = (const float*)d_in[32];
    float* outp = (float*)d_out;

    // ---- workspace carving ----
    char* ws = (char*)d_ws;
    size_t off = 0;
    auto alloc = [&](size_t bytes) -> char* {
        char* p = ws + off;
        off = (off + bytes + 255) & ~(size_t)255;
        return p;
    };
    auto BF = [&](size_t elems) { return (unsigned short*)alloc(elems * 2); };
    auto F  = [&](size_t elems) { return (float*)alloc(elems * 4); };

    unsigned short* xn_b   = BF((size_t)TOK * D);
    unsigned short* wq_t   = BF((size_t)D * D);
    unsigned short* wk_t   = BF((size_t)D * D);
    unsigned short* wv_t   = BF((size_t)D * D);
    unsigned short* wf_t   = BF((size_t)D * D);
    unsigned short* wip_t  = BF((size_t)D * 2 * D);
    unsigned short* wr1_t  = BF((size_t)D * 2 * D);
    unsigned short* wr2_t  = BF((size_t)2 * D * D);
    unsigned short* wtq_t  = BF(64 * 64);
    unsigned short* wtk_t  = BF(64 * 64);
    unsigned short* wtv_t  = BF(64 * 64);
    unsigned short* wl1_t  = BF(64 * 128);
    unsigned short* wl2_t  = BF(128 * 64);
    unsigned short* wsp_t  = BF(128 * 64);
    float*          q_f    = F((size_t)TOK * D);
    float*          kv0    = F((size_t)TOK * 2 * D);
    unsigned short* inp_b  = BF((size_t)TOK * 2 * D);
    float*          kk_f   = F((size_t)SR * 64);
    float*          vv_f   = F((size_t)SR * 64);
    float*          slots  = F((size_t)SR * 64);
    unsigned short* slotsB = BF((size_t)SR * 64);
    float*          qq_f   = F((size_t)SR * 64);
    unsigned short* hln_b  = BF((size_t)SR * 64);
    unsigned short* h1_b   = BF((size_t)SR * 128);
    float*          h2_f   = F((size_t)SR * 64);
    unsigned short* cat_b  = BF((size_t)SR * 128);
    float*          slotsP = F((size_t)SR * 64);
    float*          KV     = F((size_t)Bq * H * 64 * 64);
    float*          ksum   = F((size_t)Bq * H * 64);
    unsigned short* attn_b = BF((size_t)TOK * D);
    float*          tmp_f  = F((size_t)TOK * D);
    float*          res_f  = F((size_t)TOK * D);
    unsigned short* h2ln_b = BF((size_t)TOK * D);
    unsigned short* mlp1_b = BF((size_t)TOK * 2 * D);

    auto convT = [&](const float* W, unsigned short* Wt, int K, int Ncol) {
        size_t n = (size_t)K * Ncol;
        aid_convT_bf16<<<(unsigned)((n + 255) / 256), 256, 0, stream>>>(W, Wt, K, Ncol);
    };
    auto gemm = [&](const unsigned short* A, const unsigned short* Wt,
                    const float* bias, float* outF, int ldo, int coff,
                    unsigned short* outB, int M, int K, int Nout, int act) {
        dim3 g(M / 128, Nout / 64);
        aid_gemm_wmma<<<g, 128, 0, stream>>>(A, Wt, bias, outF, ldo, coff,
                                             outB, M, K, Nout, act);
    };
    auto ew = [&](size_t n) { return (unsigned)((n + 255) / 256); };

    // ---- weight conversion (f32 -> bf16, transposed) ----
    convT(Wq,  wq_t,  D, D);     convT(Wk,  wk_t,  D, D);
    convT(Wv,  wv_t,  D, D);     convT(Wf,  wf_t,  D, D);
    convT(Wip, wip_t, D, 2 * D); convT(Wr1, wr1_t, D, 2 * D);
    convT(Wr2, wr2_t, 2 * D, D);
    convT(Wtq, wtq_t, 64, 64);   convT(Wtk, wtk_t, 64, 64);
    convT(Wtv, wtv_t, 64, 64);
    convT(Wl1, wl1_t, 64, 128);  convT(Wl2, wl2_t, 128, 64);
    convT(Wsp, wsp_t, 128, 64);

    // ---- xn = LN1(x) ----
    aid_layernorm_bf16<<<TOK, 256, 0, stream>>>(x, ln1_g, ln1_b, xn_b, D);

    // ---- q / k / v / inp projections ----
    gemm(xn_b, wq_t,  bq,  q_f, D, 0,       nullptr, TOK, D, D, 0);
    gemm(xn_b, wk_t,  bk,  kv0, 2 * D, 0,   nullptr, TOK, D, D, 0);   // kv0[:, :1024] = k
    gemm(xn_b, wv_t,  bv,  kv0, 2 * D, D,   nullptr, TOK, D, D, 0);   // kv0[:, 1024:] = v
    gemm(xn_b, wip_t, bip, nullptr, 0, 0,   inp_b,   TOK, D, 2 * D, 0);

    // ---- kk = feat(inp @ Wtk + btk), vv = inp @ Wtv + btv ----
    gemm(inp_b, wtk_t, btk, kk_f, 64, 0, nullptr, SR, 64, 64, 2);
    gemm(inp_b, wtv_t, btv, vv_f, 64, 0, nullptr, SR, 64, 64, 0);

    // ---- slots = kv0 ----
    {
        size_t n = (size_t)SR * 64;
        aid_copy_f32_bf16<<<ew(n), 256, 0, stream>>>(kv0, slots, slotsB, n);
    }

    // ---- 3 slot-attention iterations ----
    for (int it = 0; it < 3; ++it) {
        gemm(slotsB, wtq_t, btq, qq_f, 64, 0, nullptr, SR, 64, 64, 0);
        {
            size_t n = (size_t)SR * 64;
            aid_qq_feat<<<ew(n), 256, 0, stream>>>(qq_f, kv0, n);
        }
        aid_slot_attn<<<TOK, 256, 0, stream>>>(kk_f, qq_f, vv_f,
                                               lnp_g, lnp_b, hln_b);
        gemm(hln_b, wl1_t, bl1, nullptr, 0, 0, h1_b, SR, 64, 128, 1);   // relu
        gemm(h1_b,  wl2_t, bl2, h2_f, 64, 0, nullptr, SR, 128, 64, 0);
        {
            size_t n = (size_t)SR * 64;
            aid_slots_update<<<ew(n), 256, 0, stream>>>(slots, slotsB, h2_f, n);
        }
    }

    // ---- slots = concat([kv0, slots], -1) @ Wsp + bsp ----
    {
        size_t n = (size_t)SR * 128;
        aid_concat_bf16<<<ew(n), 256, 0, stream>>>(kv0, slots, cat_b, n);
    }
    gemm(cat_b, wsp_t, bsp, slotsP, 64, 0, nullptr, SR, 128, 64, 0);

    // ---- collapsed linear attention ----
    aid_kv_accum<<<Bq * H, 256, 0, stream>>>(slotsP, KV, ksum);
    aid_attn_apply<<<TOK, 256, 0, stream>>>(q_f, KV, ksum, attn_b);

    // ---- out = x + attn @ Wf + bf ----
    gemm(attn_b, wf_t, bf, tmp_f, D, 0, nullptr, TOK, D, D, 0);
    {
        size_t n = (size_t)TOK * D;
        aid_add_f32<<<ew(n), 256, 0, stream>>>(x, tmp_f, res_f, n);
    }

    // ---- final = res + MLP(LN2(res)) ----
    aid_layernorm_bf16<<<TOK, 256, 0, stream>>>(res_f, ln2_g, ln2_b, h2ln_b, D);
    gemm(h2ln_b, wr1_t, br1, nullptr, 0, 0, mlp1_b, TOK, D, 2 * D, 1);  // relu
    gemm(mlp1_b, wr2_t, br2, tmp_f, D, 0, nullptr, TOK, 2 * D, D, 0);
    {
        size_t n = (size_t)TOK * D;
        aid_add_f32<<<ew(n), 256, 0, stream>>>(res_f, tmp_f, outp, n);
    }
}
